// AESUELOGIT_84782654423332
// MI455X (gfx1250) — matrix operand order repaired and checked by reading, not compiled, not run on hardware
//
#include <hip/hip_runtime.h>
#include <hip/hip_bf16.h>

// ---------------- problem constants ----------------
#define NDH   96        // N_DAYS * N_HOURS
#define NL    2000      // N_LINKS
#define NLP   2016      // N_LINKS padded to 63*32 for WMMA K
#define NP    20000     // N_PATHS  (= 1250*16 = 625*32)
#define NOD   4000      // N_ODS
#define KT1   63        // K tiles for GEMM1 (2016/32)
#define KS1   3         // k-split for GEMM1
#define KT1_PER 21      // 63/3
#define KT2   625       // K tiles for GEMM2 (20000/32)
#define KS2   20        // k-split for GEMM2
#define KT2_PER 32      // ceil(625/20)

#if __has_builtin(__builtin_amdgcn_sched_barrier)
#define SCHED_FENCE() __builtin_amdgcn_sched_barrier(0)
#else
#define SCHED_FENCE()
#endif

typedef __attribute__((ext_vector_type(16))) __bf16 v16bf;
typedef __attribute__((ext_vector_type(8)))  __bf16 v8bf;
typedef __attribute__((ext_vector_type(8)))  float  v8f;
typedef __attribute__((ext_vector_type(4)))  float  v4f;

// round-to-nearest-even f32 -> bf16 (pure integer ops; exact for 0/1)
static __device__ __forceinline__ __bf16 to_bf16(float f) {
  unsigned u = __float_as_uint(f);
  u += 0x7FFFu + ((u >> 16) & 1u);
  unsigned short h = (unsigned short)(u >> 16);
  return __builtin_bit_cast(__bf16, h);
}

// assemble a 16-element bf16 WMMA fragment from two 16B-aligned 8-element chunks
static __device__ __forceinline__ v16bf load_frag(const __bf16* p0, const __bf16* p1) {
  v8bf lo = *reinterpret_cast<const v8bf*>(p0);
  v8bf hi = *reinterpret_cast<const v8bf*>(p1);
  v16bf r;
#pragma unroll
  for (int i = 0; i < 8; ++i) { r[i] = lo[i]; r[8 + i] = hi[i]; }
  return r;
}

// ---------------- K0: tiled transpose+convert  D[2000][20000] f32 -> DT[20000][2016] bf16 ----------------
__global__ __launch_bounds__(256) void k_transpose(const float* __restrict__ D,
                                                   __bf16* __restrict__ DT) {
  __shared__ float tile[32][33];
  const int tx = threadIdx.x;            // 0..31
  const int ty = threadIdx.y;            // 0..7
  const int p  = blockIdx.x * 32 + tx;   // path col   (625 tiles, exact)
  const int lb = blockIdx.y * 32;        // link row   (63 tiles -> 2016)
#pragma unroll
  for (int j = 0; j < 4; ++j) {
    int li = lb + ty + j * 8;
    float v = (li < NL) ? D[(size_t)li * NP + p] : 0.0f;
    tile[ty + j * 8][tx] = v;
  }
  __syncthreads();
  const int lo = lb + tx;                // output link index (< 2016 always)
#pragma unroll
  for (int j = 0; j < 4; ++j) {
    int po = blockIdx.x * 32 + ty + j * 8;
    DT[(size_t)po * NLP + lo] = to_bf16(tile[tx][ty + j * 8]);
  }
}

// ---------------- K1: V[dh][l] = sum_f min(theta_f,0)*X[dh,l,1+f] + theta_links[l], bf16, padded ----------------
__global__ __launch_bounds__(256) void k_computeV(const float* __restrict__ X,
                                                  const float* __restrict__ theta_raw,
                                                  const float* __restrict__ theta_links,
                                                  __bf16* __restrict__ Vb) {
  int t = blockIdx.x * 256 + threadIdx.x;
  if (t >= NDH * NLP) return;
  int dh = t / NLP, l = t - dh * NLP;
  float v = 0.0f;
  if (l < NL) {
    const float* xr = X + ((size_t)dh * NL + l) * 5;
    v = theta_links[l];
#pragma unroll
    for (int f = 0; f < 4; ++f) v += fminf(theta_raw[f], 0.0f) * xr[1 + f];
  }
  Vb[t] = to_bf16(v);
}

// ---------------- GEMM1 helpers ----------------
static __device__ __forceinline__ void g1_load(const __bf16* __restrict__ arow,
                                               const __bf16* __restrict__ bbase,
                                               int kb, int half, v16bf& a, v16bf* b) {
  a = load_frag(arow + kb + half * 8, arow + kb + 16 + half * 8);
#pragma unroll
  for (int nt = 0; nt < 6; ++nt) {
    const __bf16* br = bbase + (size_t)(nt * 16) * NLP + kb;
    b[nt] = load_frag(br + half * 8, br + 16 + half * 8);
  }
}
static __device__ __forceinline__ void g1_mma(v8f* acc, const v16bf& a, const v16bf* b) {
#pragma unroll
  for (int nt = 0; nt < 6; ++nt)
    acc[nt] = __builtin_amdgcn_wmma_f32_16x16x32_bf16(
        false, a, false, b[nt], (short)0, acc[nt], false, false);
}

// ---------------- K2: GEMM1  vfTp[ky][p][dh] = sum_{l in ky} DT[p][l] * V[dh][l] ----------------
__global__ __launch_bounds__(32) void k_gemm1(const __bf16* __restrict__ DT,
                                              const __bf16* __restrict__ Vb,
                                              float* __restrict__ vfTp) {
  const int lane = threadIdx.x;
  const int m0   = blockIdx.x * 16;     // path strip
  const int mrow = lane & 15;
  const int half = lane >> 4;
  v8f z = {0.f,0.f,0.f,0.f,0.f,0.f,0.f,0.f};
  v8f acc[6];
#pragma unroll
  for (int nt = 0; nt < 6; ++nt) acc[nt] = z;

  const __bf16* arow  = DT + (size_t)(m0 + mrow) * NLP;
  const __bf16* bbase = Vb + (size_t)mrow * NLP;

  const int ktB = blockIdx.y * KT1_PER;
  const int ktE = ktB + KT1_PER;
  v16bf a0, a1, b0[6], b1[6];
  g1_load(arow, bbase, ktB * 32, half, a0, b0);
  int i = ktB;
  for (; i + 2 <= ktE; i += 2) {
    g1_load(arow, bbase, (i + 1) * 32, half, a1, b1);
    SCHED_FENCE();
    g1_mma(acc, a0, b0);
    if (i + 2 < ktE) { g1_load(arow, bbase, (i + 2) * 32, half, a0, b0); SCHED_FENCE(); }
    g1_mma(acc, a1, b1);
  }
  if (i < ktE) g1_mma(acc, a0, b0);

  const int n = lane & 15, mh = half * 8;
  float* outp = vfTp + (size_t)blockIdx.y * NP * NDH;
#pragma unroll
  for (int nt = 0; nt < 6; ++nt)
#pragma unroll
    for (int r = 0; r < 8; ++r)
      outp[(size_t)(m0 + mh + r) * NDH + nt * 16 + n] = acc[nt][r];
}

// ---------------- K2b: reduce GEMM1 k-split ----------------
__global__ __launch_bounds__(256) void k_reduce_vf(const float* __restrict__ vfTp,
                                                   float* __restrict__ vfT) {
  int t = blockIdx.x * 256 + threadIdx.x;
  if (t >= NP * NDH) return;
  float s = 0.0f;
#pragma unroll
  for (int j = 0; j < KS1; ++j) s += vfTp[(size_t)j * NP * NDH + t];
  vfT[t] = s;
}

// ---------------- K3: segmented softmax + flow  ->  fT[dh][p] bf16 ----------------
static __device__ __forceinline__ int lower_bound(const int* a, int n, int key) {
  int lo = 0, hi = n;
  while (lo < hi) { int mid = (lo + hi) >> 1; if (a[mid] < key) lo = mid + 1; else hi = mid; }
  return lo;
}

__global__ __launch_bounds__(256) void k_softmax(const float* __restrict__ vfT,
                                                 const int* __restrict__ od_of_path,
                                                 const float* __restrict__ q_sqrt,
                                                 __bf16* __restrict__ fT) {
  int t = blockIdx.x * 256 + threadIdx.x;
  if (t >= NOD * NDH) return;
  const int od = t / NDH, dh = t - od * NDH;   // consecutive threads: same od, consecutive dh
  const int s = lower_bound(od_of_path, NP, od);
  const int e = lower_bound(od_of_path, NP, od + 1);
  if (s >= e) return;
  float m = -1e30f;
  for (int p = s; p < e; ++p) m = fmaxf(m, vfT[(size_t)p * NDH + dh]);
  float sum = 0.0f;
  for (int p = s; p < e; ++p) sum += __expf(vfT[(size_t)p * NDH + dh] - m);
  const float q = q_sqrt[od] * q_sqrt[od];
  const float scale = q / sum;
  for (int p = s; p < e; ++p)
    fT[(size_t)dh * NP + p] = to_bf16(__expf(vfT[(size_t)p * NDH + dh] - m) * scale);
}

// ---------------- GEMM2 helpers (A: raw f32 staged, converted in compute phase) ----------------
struct RawA { v4f c[4]; };
static __device__ __forceinline__ void g2_load(const float* __restrict__ arow,
                                               const __bf16* __restrict__ bbase,
                                               int kb, int half, RawA& ra, v16bf* b) {
  const float* p0 = arow + kb + half * 8;
  ra.c[0] = *reinterpret_cast<const v4f*>(p0);
  ra.c[1] = *reinterpret_cast<const v4f*>(p0 + 4);
  ra.c[2] = *reinterpret_cast<const v4f*>(p0 + 16);
  ra.c[3] = *reinterpret_cast<const v4f*>(p0 + 20);
#pragma unroll
  for (int nt = 0; nt < 6; ++nt) {
    const __bf16* br = bbase + (size_t)(nt * 16) * NP + kb;
    b[nt] = load_frag(br + half * 8, br + 16 + half * 8);
  }
}
static __device__ __forceinline__ void g2_mma(v8f* acc, const RawA& ra, const v16bf* b) {
  v16bf a;
#pragma unroll
  for (int j = 0; j < 4; ++j)
#pragma unroll
    for (int i = 0; i < 4; ++i) a[j * 4 + i] = to_bf16(ra.c[j][i]);
#pragma unroll
  for (int nt = 0; nt < 6; ++nt)
    acc[nt] = __builtin_amdgcn_wmma_f32_16x16x32_bf16(
        false, a, false, b[nt], (short)0, acc[nt], false, false);
}

// ---------------- K4: GEMM2  xTp[ky][l][dh] = sum_{p in ky} D[l][p] * f[p][dh] ----------------
__global__ __launch_bounds__(32) void k_gemm2(const float* __restrict__ D,
                                              const __bf16* __restrict__ fT,
                                              float* __restrict__ xTp) {
  const int lane = threadIdx.x;
  const int l0   = blockIdx.x * 16;     // link strip (125 tiles, exact)
  const int ktB  = blockIdx.y * KT2_PER;
  const int ktE  = (ktB + KT2_PER < KT2) ? (ktB + KT2_PER) : KT2;
  const int mrow = lane & 15;
  const int half = lane >> 4;
  v8f z = {0.f,0.f,0.f,0.f,0.f,0.f,0.f,0.f};
  v8f acc[6];
#pragma unroll
  for (int nt = 0; nt < 6; ++nt) acc[nt] = z;

  const float*  arow  = D  + (size_t)(l0 + mrow) * NP;
  const __bf16* bbase = fT + (size_t)mrow * NP;

  RawA ra0, ra1;
  v16bf b0[6], b1[6];
  g2_load(arow, bbase, ktB * 32, half, ra0, b0);
  int i = ktB;
  for (; i + 2 <= ktE; i += 2) {
    g2_load(arow, bbase, (i + 1) * 32, half, ra1, b1);
    SCHED_FENCE();
    g2_mma(acc, ra0, b0);
    if (i + 2 < ktE) { g2_load(arow, bbase, (i + 2) * 32, half, ra0, b0); SCHED_FENCE(); }
    g2_mma(acc, ra1, b1);
  }
  if (i < ktE) g2_mma(acc, ra0, b0);

  const int n = lane & 15, mh = half * 8;
  float* outp = xTp + (size_t)blockIdx.y * NL * NDH;
#pragma unroll
  for (int nt = 0; nt < 6; ++nt)
#pragma unroll
    for (int r = 0; r < 8; ++r)
      outp[(size_t)(l0 + mh + r) * NDH + nt * 16 + n] = acc[nt][r];
}

// ---------------- K5: reduce k-split + BPR finish ----------------
__global__ __launch_bounds__(256) void k_final(const float* __restrict__ X,
                                               const float* __restrict__ log_alpha,
                                               const float* __restrict__ beta_raw,
                                               const float* __restrict__ kcap,
                                               const float* __restrict__ xTp,
                                               float* __restrict__ out) {
  int t = blockIdx.x * 256 + threadIdx.x;
  if (t >= NDH * NL) return;
  const int dh = t / NL, l = t - dh * NL;
  float x = 0.0f;
#pragma unroll
  for (int j = 0; j < KS2; ++j) x += xTp[(size_t)j * NL * NDH + (size_t)l * NDH + dh];
  x = fmaxf(x, 0.0f);
  const float tt    = X[((size_t)dh * NL + l) * 5];
  const float alpha = __expf(log_alpha[l]);
  float beta = beta_raw[l];
  beta = fminf(fmaxf(beta, 1e-12f), 4.0f);
  out[t] = tt * (1.0f + alpha * powf(x / kcap[l], beta));
}

// ---------------- launcher ----------------
extern "C" void kernel_launch(void* const* d_in, const int* in_sizes, int n_in,
                              void* d_out, int out_size, void* d_ws, size_t ws_size,
                              hipStream_t stream) {
  (void)in_sizes; (void)n_in; (void)out_size; (void)ws_size;
  const float* X           = (const float*)d_in[0];
  const float* theta_raw   = (const float*)d_in[1];
  const float* theta_links = (const float*)d_in[2];
  const float* q_sqrt      = (const float*)d_in[3];
  const float* log_alpha   = (const float*)d_in[4];
  const float* beta_raw    = (const float*)d_in[5];
  const float* kcap        = (const float*)d_in[6];
  const float* D           = (const float*)d_in[7];
  const int*   od_of_path  = (const int*)d_in[8];
  float*       out         = (float*)d_out;

  // workspace carve-up (all 256B aligned by construction)
  char* ws = (char*)d_ws;
  __bf16* DT   = (__bf16*)(ws);                      // 20000*2016*2      = 80,640,000
  __bf16* Vb   = (__bf16*)(ws +  80640000);          //    96*2016*2     =    387,072
  float*  vfTp = (float*) (ws +  81027072);          // 3*20000*96*4     = 23,040,000
  float*  vfT  = (float*) (ws + 104067072);          //   20000*96*4     =  7,680,000
  __bf16* fT   = (__bf16*)(ws + 111747072);          //   96*20000*2     =  3,840,000
  float*  xTp  = (float*) (ws + 115587072);          // 20*2000*96*4     = 15,360,000
                                                     // total            = 130,947,072

  k_transpose<<<dim3(NP / 32, NLP / 32), dim3(32, 8), 0, stream>>>(D, DT);
  k_computeV<<<(NDH * NLP + 255) / 256, 256, 0, stream>>>(X, theta_raw, theta_links, Vb);
  k_gemm1<<<dim3(NP / 16, KS1), 32, 0, stream>>>(DT, Vb, vfTp);
  k_reduce_vf<<<(NP * NDH + 255) / 256, 256, 0, stream>>>(vfTp, vfT);
  k_softmax<<<(NOD * NDH + 255) / 256, 256, 0, stream>>>(vfT, od_of_path, q_sqrt, fT);
  k_gemm2<<<dim3(NL / 16, KS2), 32, 0, stream>>>(D, fT, xTp);
  k_final<<<(NDH * NL + 255) / 256, 256, 0, stream>>>(X, log_alpha, beta_raw, kcap, xTp, out);
}
